// TopKGate_25872882992016
// MI455X (gfx1250) — matrix operand order repaired and checked by reading, not compiled, not run on hardware
//
#include <hip/hip_runtime.h>

// CDNA5 / gfx1250 wave32 WMMA types
typedef __attribute__((ext_vector_type(16))) __bf16 v16bf;
typedef __attribute__((ext_vector_type(8)))  float  v8f;

#define KDIM 4096
#define NEXP 64
#define NTOK 8192
#define MT   2          // 16-token M-tiles per wave (B-register reuse)

// Pack 16 fp32 -> 16 bf16 in WMMA register element order.
__device__ __forceinline__ v16bf cvt16(const float4 f0, const float4 f1,
                                       const float4 f2, const float4 f3) {
  v16bf r;
  r[0]  = (__bf16)f0.x; r[1]  = (__bf16)f0.y; r[2]  = (__bf16)f0.z; r[3]  = (__bf16)f0.w;
  r[4]  = (__bf16)f1.x; r[5]  = (__bf16)f1.y; r[6]  = (__bf16)f1.z; r[7]  = (__bf16)f1.w;
  r[8]  = (__bf16)f2.x; r[9]  = (__bf16)f2.y; r[10] = (__bf16)f2.z; r[11] = (__bf16)f2.w;
  r[12] = (__bf16)f3.x; r[13] = (__bf16)f3.y; r[14] = (__bf16)f3.z; r[15] = (__bf16)f3.w;
  return r;
}

__global__ __launch_bounds__(256)
void topk_gate_wmma(const float* __restrict__ x, const float* __restrict__ W,
                    float* __restrict__ out) {
  const int  lane   = threadIdx.x & 31;
  const int  lane16 = lane & 15;
  const bool hi     = lane >= 16;
  const int  wave   = threadIdx.x >> 5;
  const int  tokenBase = (blockIdx.x * (blockDim.x >> 5) + wave) * (16 * MT);

  // Per-lane source rows.
  const float* xr[MT];
#pragma unroll
  for (int m = 0; m < MT; ++m)
    xr[m] = x + (size_t)(tokenBase + m * 16 + lane16) * KDIM;
  const float* wr[4];
#pragma unroll
  for (int t = 0; t < 4; ++t)
    wr[t] = W + (size_t)(t * 16 + lane16) * KDIM;

  // A-matrix K sub-pattern per lane-half: {0..7,16..23} vs {8..15,24..31}
  const int aoff = hi ? 8 : 0;
  // B-matrix: same K sub-pattern as A: {0..7,16..23} vs {8..15,24..31}
  const int boff = hi ? 8 : 0;

  v8f acc[MT][4];
#pragma unroll
  for (int m = 0; m < MT; ++m)
#pragma unroll
    for (int t = 0; t < 4; ++t)
      acc[m][t] = (v8f){};

  for (int kb = 0; kb < KDIM; kb += 32) {
    v16bf a[MT];
#pragma unroll
    for (int m = 0; m < MT; ++m) {
      const float4* p = reinterpret_cast<const float4*>(xr[m] + kb + aoff);
      // floats at +0..7 and +16..23 relative to (kb+aoff)
      a[m] = cvt16(p[0], p[1], p[4], p[5]);
    }
#pragma unroll
    for (int t = 0; t < 4; ++t) {
      const float4* q = reinterpret_cast<const float4*>(wr[t] + kb + boff);
      v16bf b = cvt16(q[0], q[1], q[4], q[5]);
#pragma unroll
      for (int m = 0; m < MT; ++m)
        acc[m][t] = __builtin_amdgcn_wmma_f32_16x16x32_bf16(
            false, a[m], false, b, (short)0, acc[m][t], false, false);
    }
  }

  float* wout = out;                         // [NTOK * NEXP] fp32 dense gate weights
  float* iout = out + (size_t)NTOK * NEXP;   // [NTOK * 2] indices (as exact floats)

#pragma unroll
  for (int m = 0; m < MT; ++m) {
#pragma unroll
    for (int j = 0; j < 8; ++j) {
      // This lane's 4 logits for token row (M = j or j+8 per lane half).
      float v[4];
#pragma unroll
      for (int t = 0; t < 4; ++t) v[t] = acc[m][t][j];

      // Local top-2 (value, expert index) among this lane's 4 experts.
      float av = v[0]; int ai = lane16;
      float bv = v[1]; int bi = lane16 + 16;
      if (bv > av) { float tv = av; av = bv; bv = tv; int ti = ai; ai = bi; bi = ti; }
#pragma unroll
      for (int t = 2; t < 4; ++t) {
        float cv = v[t]; int ci = lane16 + 16 * t;
        if (cv > av)      { bv = av; bi = ai; av = cv; ai = ci; }
        else if (cv > bv) { bv = cv; bi = ci; }
      }

      // Butterfly merge across the 16-lane half (masks 1,2,4,8 stay in-half).
#pragma unroll
      for (int mask = 1; mask <= 8; mask <<= 1) {
        float oav = __shfl_xor(av, mask, 32);
        int   oai = __shfl_xor(ai, mask, 32);
        float obv = __shfl_xor(bv, mask, 32);
        int   obi = __shfl_xor(bi, mask, 32);
        if (oav > av) {
          float nbv; int nbi;
          if (av > obv) { nbv = av; nbi = ai; } else { nbv = obv; nbi = obi; }
          av = oav; ai = oai; bv = nbv; bi = nbi;
        } else if (oav > bv) {
          bv = oav; bi = oai;
        }
      }

      // Softmax: av is the global max over 64 experts; sum exp across half-wave.
      float p[4];
      float se = 0.0f;
#pragma unroll
      for (int t = 0; t < 4; ++t) { p[t] = __expf(v[t] - av); se += p[t]; }
#pragma unroll
      for (int mask = 1; mask <= 8; mask <<= 1)
        se += __shfl_xor(se, mask, 32);
      const float inv = 1.0f / se;

      const int tok = tokenBase + m * 16 + j + (hi ? 8 : 0);
      float* row = wout + (size_t)tok * NEXP;
#pragma unroll
      for (int t = 0; t < 4; ++t) {
        const int e = lane16 + 16 * t;
        row[e] = (e == ai || e == bi) ? p[t] * inv : 0.0f;
      }
      if (lane16 == 0) {
        iout[(size_t)tok * 2 + 0] = (float)ai;
        iout[(size_t)tok * 2 + 1] = (float)bi;
      }
    }
  }
}

extern "C" void kernel_launch(void* const* d_in, const int* in_sizes, int n_in,
                              void* d_out, int out_size, void* d_ws, size_t ws_size,
                              hipStream_t stream) {
  const float* x = (const float*)d_in[0];   // [8192, 4096] fp32
  const float* W = (const float*)d_in[1];   // [64, 4096]   fp32
  float* out = (float*)d_out;               // weights [8192*64] ++ top_i [8192*2]

  const int waves_per_block  = 256 / 32;
  const int tokens_per_block = waves_per_block * 16 * MT;   // 256
  dim3 block(256);
  dim3 grid(NTOK / tokens_per_block);                       // 32 blocks

  topk_gate_wmma<<<grid, block, 0, stream>>>(x, W, out);
}
